// ForceModule_34136400068648
// MI455X (gfx1250) — matrix operand order repaired and checked by reading, not compiled, not run on hardware
//
#include <hip/hip_runtime.h>

// ---------------------------------------------------------------------------
// MI455X (gfx1250) fused GNN energy kernel.
//   energy = KJ2KCAL * sum( relu([h | phi@h] @ W1) @ w2 )
// phi is never materialized: each wave computes its 16x32 f16 A-fragment of
// exp(-d) on the fly and feeds v_wmma_f32_16x16x32_f16 directly.
// Round-2 refinements:
//   * batch all 8 B-fragment loads per K-step before the WMMA block so the
//     exp/sqrt VALU work overlaps the L2 load latency (no per-WMMA drain)
//   * skip the 8 WMMAs for all-zero phi tiles
//     via a wave-uniform ballot branch (EXEC stays all-1s as WMMA requires)
// ---------------------------------------------------------------------------

typedef __attribute__((ext_vector_type(16))) _Float16 v16h;
typedef __attribute__((ext_vector_type(8)))  _Float16 v8h;
typedef __attribute__((ext_vector_type(8)))  float    v8f;

#define NM2ANG   10.0f
#define KJ2KCAL  0.2390057361376673f
#define CUTOFF   5.0f
#define EPSF     1e-12f

static constexpr int kD = 128;    // embedding dim
static constexpr int kH = 256;    // hidden dim
static constexpr int kF = 2 * kD; // feat dim = 256

// ------------------------------ prep kernels -------------------------------

__global__ void k_zero_out(float* out) { out[0] = 0.0f; }

__global__ void k_prep_pos(const float* __restrict__ pos_in,
                           float* __restrict__ pos_s,
                           float* __restrict__ sq, int N) {
    int n = blockIdx.x * blockDim.x + threadIdx.x;
    if (n >= N) return;
    float x = NM2ANG * pos_in[3 * n + 0];
    float y = NM2ANG * pos_in[3 * n + 1];
    float z = NM2ANG * pos_in[3 * n + 2];
    pos_s[3 * n + 0] = x;
    pos_s[3 * n + 1] = y;
    pos_s[3 * n + 2] = z;
    sq[n] = x * x + y * y + z * z;
}

// h = emb[node_types]: store row-major f16 into feat[:, 0:128] and transposed
// f16 into hT[128][N] (so WMMA B fragments are contiguous 32B loads).
__global__ void k_prep_h(const int* __restrict__ types,
                         const float* __restrict__ emb,
                         _Float16* __restrict__ hT,
                         _Float16* __restrict__ feat, int N) {
    int idx = blockIdx.x * blockDim.x + threadIdx.x;
    if (idx >= N * kD) return;
    int n = idx >> 7;        // / 128
    int c = idx & (kD - 1);  // % 128
    float v = emb[types[n] * kD + c];
    _Float16 hv = (_Float16)v;
    hT[(size_t)c * N + n] = hv;
    feat[(size_t)n * kF + c] = hv;
}

// W1T[h][k] = W1[k][h], f16
__global__ void k_prep_w1(const float* __restrict__ W1,
                          _Float16* __restrict__ w1t) {
    int idx = blockIdx.x * blockDim.x + threadIdx.x;
    if (idx >= kF * kH) return;
    int k = idx >> 8;        // row in W1 (0..255)
    int h = idx & (kH - 1);  // col in W1 (0..255)
    w1t[(size_t)h * kF + k] = (_Float16)W1[idx];
}

// ------------------------- phi @ h  (the big GEMM) -------------------------
// grid.x = N/16 row tiles; block = 256 threads = 8 waves.
// Wave w accumulates over K-blocks jb = (w + 8*t)*32; cross-wave reduce in LDS.
__global__ void __launch_bounds__(256)
k_phi_gemm(const float* __restrict__ pos_s, const float* __restrict__ sq,
           const _Float16* __restrict__ hT, _Float16* __restrict__ feat, int N) {
    const int i0   = blockIdx.x * 16;
    const int tid  = threadIdx.x;
    const int wav  = tid >> 5;        // 0..7
    const int lane = tid & 31;
    const int l15  = lane & 15;
    const int hi   = (lane >> 4) & 1; // lane-half

    // this lane's output row (A-fragment: both lane-halves map M=0..15)
    const int i = i0 + l15;
    const float xi = pos_s[3 * i + 0];
    const float yi = pos_s[3 * i + 1];
    const float zi = pos_s[3 * i + 2];
    const float sqi = sq[i];

    const v8f vzero = {0.f, 0.f, 0.f, 0.f, 0.f, 0.f, 0.f, 0.f};
    v8f c[kD / 16];
#pragma unroll
    for (int t = 0; t < kD / 16; ++t) c[t] = vzero;

    // per-lane B base pointers (row = column of h; fixed for whole kernel)
    const _Float16* brow[kD / 16];
#pragma unroll
    for (int ct = 0; ct < kD / 16; ++ct)
        brow[ct] = hT + (size_t)(ct * 16 + l15) * N + (hi ? 16 : 0);

    const int kbase = hi ? 16 : 0;    // A-fragment K base for this lane-half
    const int nks = N / 32;

    for (int ks = wav; ks < nks; ks += 8) {
        const int jb = ks * 32;

        // ---- issue all 8 B-fragment loads first (16x global_load_b128) ----
        v16h b[kD / 16];
#pragma unroll
        for (int ct = 0; ct < kD / 16; ++ct)
            b[ct] = *(const v16h*)(brow[ct] + jb);

        __builtin_prefetch(pos_s + 3 * ((jb + 256) & (N - 1)), 0, 0);

        // ---- build 16x32 f16 A fragment of phi while loads are in flight --
        // lane-half 0: elements 0..15 -> K=0..15
        // lane-half 1: elements 0..15 -> K=16..31
        v16h a;
        bool any = false;
#pragma unroll
        for (int e = 0; e < 16; ++e) {
            const int k = kbase + e;
            const int j = jb + k;
            const float xj = pos_s[3 * j + 0];
            const float yj = pos_s[3 * j + 1];
            const float zj = pos_s[3 * j + 2];
            const float d2 = sqi + sq[j] - 2.0f * (xi * xj + yi * yj + zi * zj);
            const float d  = sqrtf(fmaxf(d2, EPSF));
            const bool  on = (d2 > EPSF) && (d <= CUTOFF);
            any |= on;
            const float phi = on ? __expf(-d) : 0.0f;
            a[e] = (_Float16)phi;
        }

        // ---- wave-uniform skip of all-zero tiles (EXEC untouched) ----
        if (__builtin_amdgcn_ballot_w32(any) != 0u) {
#pragma unroll
            for (int ct = 0; ct < kD / 16; ++ct)
                c[ct] = __builtin_amdgcn_wmma_f32_16x16x32_f16(
                    false, a, false, b[ct], (short)0, c[ct], false, false);
        }
    }

    // ---- cross-wave reduction of partial 16x128 tiles via LDS (64 KB) ----
    __shared__ float red[8][16][kD];
#pragma unroll
    for (int ct = 0; ct < kD / 16; ++ct) {
#pragma unroll
        for (int r = 0; r < 8; ++r) {
            const int m = r + (hi ? 8 : 0);       // C/D layout: row = r + 8*half
            red[wav][m][ct * 16 + l15] = c[ct][r];
        }
    }
    __syncthreads();

    for (int e = tid; e < 16 * kD; e += 256) {
        const int m = e >> 7;
        const int col = e & (kD - 1);
        float s = 0.0f;
#pragma unroll
        for (int w = 0; w < 8; ++w) s += red[w][m][col];
        feat[(size_t)(i0 + m) * kF + kD + col] = (_Float16)s;
    }
}

// --------------------- MLP: relu(feat @ W1) @ w2, reduced -------------------
// grid.x = N/16 row tiles; block = 128 threads = 4 waves; wave owns 4 H-tiles.
__global__ void __launch_bounds__(128)
k_mlp(const _Float16* __restrict__ feat, const _Float16* __restrict__ w1t,
      const float* __restrict__ w2, float* __restrict__ out, int N) {
    const int i0   = blockIdx.x * 16;
    const int tid  = threadIdx.x;
    const int wav  = tid >> 5;        // 0..3
    const int lane = tid & 31;
    const int l15  = lane & 15;
    const int hi   = (lane >> 4) & 1;
    const int i = i0 + l15;

    const v8f vzero = {0.f, 0.f, 0.f, 0.f, 0.f, 0.f, 0.f, 0.f};
    float eacc = 0.0f;

#pragma unroll
    for (int nt4 = 0; nt4 < 4; ++nt4) {
        const int n0 = (wav * 4 + nt4) * 16;
        const float w2v = w2[n0 + l15];            // C layout: lane -> column
        v8f c = vzero;
#pragma unroll
        for (int ks = 0; ks < kF / 32; ++ks) {
            const int k0 = ks * 32;
            // A fragment from row-major feat: two contiguous v8h chunks
            const _Float16* ap = feat + (size_t)i * kF + k0 + (hi ? 16 : 0);
            v8h alo = *(const v8h*)ap;
            v8h ahi = *(const v8h*)(ap + 8);
            v16h a = __builtin_shufflevector(alo, ahi, 0, 1, 2, 3, 4, 5, 6, 7,
                                             8, 9, 10, 11, 12, 13, 14, 15);
            // B fragment from transposed W1: contiguous 16 K-values
            const _Float16* bp =
                w1t + (size_t)(n0 + l15) * kF + k0 + (hi ? 16 : 0);
            v16h b = *(const v16h*)bp;
            c = __builtin_amdgcn_wmma_f32_16x16x32_f16(
                false, a, false, b, (short)0, c, false, false);
        }
#pragma unroll
        for (int r = 0; r < 8; ++r) {
            float v = c[r];
            v = v > 0.0f ? v : 0.0f;               // relu
            eacc += v * w2v;
        }
    }

    // wave32 reduction, then one atomic per wave (pre-scaled to kcal/mol)
    for (int off = 16; off > 0; off >>= 1) eacc += __shfl_down(eacc, off, 32);
    if (lane == 0) atomicAdd(out, eacc * KJ2KCAL);
}

// ------------------------------- launcher ----------------------------------

extern "C" void kernel_launch(void* const* d_in, const int* in_sizes, int n_in,
                              void* d_out, int out_size, void* d_ws, size_t ws_size,
                              hipStream_t stream) {
    const float* positions  = (const float*)d_in[0];  // [N,3] f32 (nm)
    const int*   node_types = (const int*)d_in[1];    // [N] i32
    const float* emb        = (const float*)d_in[2];  // [T,128] f32
    const float* W1         = (const float*)d_in[3];  // [256,256] f32
    const float* w2         = (const float*)d_in[4];  // [256] f32
    float* out = (float*)d_out;                       // scalar f32

    const int N = in_sizes[1];                        // 8192

    // workspace layout (all 256B-aligned by construction)
    char* ws = (char*)d_ws;
    float*     pos_s = (float*)ws;                                    // 3N f32
    float*     sq    = (float*)(ws + (size_t)N * 3 * 4);              // N f32
    _Float16*  hT    = (_Float16*)(ws + (size_t)N * 4 * 4);           // [128][N] f16
    _Float16*  feat  = (_Float16*)(ws + (size_t)N * 4 * 4 +
                                   (size_t)kD * N * 2);               // [N][256] f16
    _Float16*  w1t   = (_Float16*)(ws + (size_t)N * 4 * 4 +
                                   (size_t)kD * N * 2 +
                                   (size_t)N * kF * 2);               // [256][256] f16

    k_zero_out<<<1, 1, 0, stream>>>(out);
    k_prep_pos<<<(N + 255) / 256, 256, 0, stream>>>(positions, pos_s, sq, N);
    k_prep_h<<<(N * kD + 255) / 256, 256, 0, stream>>>(node_types, emb, hT, feat, N);
    k_prep_w1<<<(kF * kH + 255) / 256, 256, 0, stream>>>(W1, w1t);
    k_phi_gemm<<<N / 16, 256, 0, stream>>>(pos_s, sq, hT, feat, N);
    k_mlp<<<N / 16, 128, 0, stream>>>(feat, w1t, w2, out, N);
}
